// Lfm2ShortConv_32762010534210
// MI455X (gfx1250) — matrix operand order repaired and checked by reading, not compiled, not run on hardware
//
#include <hip/hip_runtime.h>
#include <hip/hip_bf16.h>

// ---------------- problem constants (from reference) ----------------
#define TDIM 8192
#define HDIM 2048
#define KCONV 4

// ---------------- types ----------------
typedef __attribute__((ext_vector_type(16))) __bf16 v16bf;
typedef __attribute__((ext_vector_type(8)))  float  v8f;
typedef __attribute__((ext_vector_type(4)))  unsigned int u32x4;

union FragAB {
    u32x4 u[2];
    v16bf v;
};

__device__ __forceinline__ unsigned short f2bf(float f) {
    // round-to-nearest-even f32 -> bf16
    unsigned int u = __float_as_uint(f);
    unsigned int r = u + 0x7FFFu + ((u >> 16) & 1u);
    return (unsigned short)(r >> 16);
}
__device__ __forceinline__ float bf2f(unsigned short h) {
    return __uint_as_float(((unsigned int)h) << 16);
}

// ---------------- CDNA5 async global->LDS copy (ASYNCcnt-tracked) ----------------
// ISA 08_async_tensor: GLOBAL_LOAD_ASYNC_TO_LDS_B128, GV mode (saddr = off/NULL).
// VDST VGPR holds the LDS byte offset; the low
// 32 bits of a generic pointer to __shared__ memory are exactly that offset.
__device__ __forceinline__ void async_copy_b128(void* lds_dst, const void* gsrc) {
    unsigned loff = (unsigned)(unsigned long long)lds_dst;
    asm volatile("global_load_async_to_lds_b128 %0, %1, off"
                 :: "v"(loff), "v"(gsrc) : "memory");
}
__device__ __forceinline__ void wait_async0() {
    asm volatile("s_wait_asynccnt 0x0" ::: "memory");
}

__device__ __forceinline__ void load_bfrag(FragAB& b, const unsigned short* bp) {
    b.u[0] = *(const u32x4*)bp;
    b.u[1] = *(const u32x4*)(bp + 16);
}

// scheduler pipeline directives: 0x100 = DS-read group, 0x008 = MFMA/WMMA group
#define SGB(mask, size) __builtin_amdgcn_sched_group_barrier((mask), (size), 0)

// ---------------- conversion kernels ----------------
__global__ void cvt_f32_to_bf16(const float* __restrict__ src,
                                unsigned short* __restrict__ dst, int n) {
    int i = blockIdx.x * blockDim.x + threadIdx.x;
    if (i < n) dst[i] = f2bf(src[i]);
}

// src [rows][cols] f32 -> dst [cols][rows] bf16
__global__ void transpose_to_bf16(const float* __restrict__ src,
                                  unsigned short* __restrict__ dst,
                                  int rows, int cols) {
    int c = blockIdx.x * blockDim.x + threadIdx.x;
    int r = blockIdx.y;
    if (c < cols && r < rows)
        dst[(size_t)c * rows + r] = f2bf(src[(size_t)r * cols + c]);
}

// ---------------- GEMM tiling params ----------------
#define TM 64
#define TN 128
#define TK 32
#define LDA 40   // padded ushort stride for A tile (conflict-free, 16B-aligned rows)
#define LDB 32   // unpadded B tile stride (keeps double-buffered LDS < 64KB)

// GEMM1: BCx = X @ W_in + b_in ; emit Bx = B*x (f32) and C (bf16)
__global__ void __launch_bounds__(256)
gemm1_kernel(const unsigned short* __restrict__ Xb,    // [T][H] bf16
             const unsigned short* __restrict__ WinT,  // [3H][H] bf16 (N-major)
             const float* __restrict__ b_in,           // [3H]
             float* __restrict__ Bx,                   // [T][H] f32
             unsigned short* __restrict__ Cg)          // [T][H] bf16
{
    __shared__ unsigned short As[2][TM * LDA];
    __shared__ unsigned short Bs[2][3 * TN * LDB];

    const int tid   = threadIdx.x;
    const int lane  = tid & 31;
    const int wave  = tid >> 5;           // 0..7
    const int wm    = wave >> 1;          // 0..3  (M sub-tile)
    const int wn    = wave & 1;           // 0..1  (N half)
    const int khalf = lane >> 4;          // 0/1
    const int l15   = lane & 15;
    const int t0    = blockIdx.y * TM;
    const int n0    = blockIdx.x * TN;

    v8f accB[4], accX[4], accC[4];
#pragma unroll
    for (int s = 0; s < 4; ++s) { accB[s] = 0.0f; accX[s] = 0.0f; accC[s] = 0.0f; }

    // async-stage one K-slab (A: 64x64B, B: 3 x 128x64B) into LDS stage s
    auto stage_load = [&](int k0, int s) {
        {
            int row = tid >> 2, ch = tid & 3;
            async_copy_b128(&As[s][row * LDA + ch * 8],
                            Xb + (size_t)(t0 + row) * HDIM + k0 + ch * 8);
        }
#pragma unroll
        for (int rep = 0; rep < 6; ++rep) {            // 3*128*4 = 1536 chunks / 256 thr
            int i = rep * 256 + tid;
            int g = i >> 9, rem = i & 511;
            int row = rem >> 2, ch = rem & 3;
            async_copy_b128(&Bs[s][(g * TN + row) * LDB + ch * 8],
                            WinT + (size_t)(g * HDIM + n0 + row) * HDIM + k0 + ch * 8);
        }
    };

    stage_load(0, 0);
    wait_async0();
    __syncthreads();

    // fragment index idx = sub*3 + g  ->  LDS address of B fragment
    auto bfrag_addr = [&](int idx, int cur) -> const unsigned short* {
        const int sub = idx / 3, g = idx % 3;
        const int n = wn * 64 + sub * 16 + l15;
        return &Bs[cur][(g * TN + n) * LDB + khalf * 8];
    };

    const int NSTEP = HDIM / TK;
    for (int ks = 0; ks < NSTEP; ++ks) {
        const int cur = ks & 1;
        if (ks + 1 < NSTEP) stage_load((ks + 1) * TK, cur ^ 1);

        // A fragment (16x32 bf16: VGPR0-3 K=8*khalf.., VGPR4-7 K=16+8*khalf..)
        FragAB a;
        {
            const unsigned short* ap = &As[cur][(wm * 16 + l15) * LDA + khalf * 8];
            a.u[0] = *(const u32x4*)ap;
            a.u[1] = *(const u32x4*)(ap + 16);
        }
        // depth-2 software pipeline over 3 rotating B-fragment buffers:
        // slot i = [WMMA_i, load frag_{i+2}] -> WMMA_i waits dscnt<=2, with two
        // WMMA executions of LDS latency cover per fragment.
        FragAB bf[3];
        load_bfrag(bf[0], bfrag_addr(0, cur));
        load_bfrag(bf[1], bfrag_addr(1, cur));
#pragma unroll
        for (int idx = 0; idx < 12; ++idx) {
            const int sub = idx / 3, g = idx % 3;
            if (idx + 2 < 12)
                load_bfrag(bf[(idx + 2) % 3], bfrag_addr(idx + 2, cur));
            const v16bf bv = bf[idx % 3].v;
            if (g == 0)
                accB[sub] = __builtin_amdgcn_wmma_f32_16x16x32_bf16(
                    false, a.v, false, bv, (short)0, accB[sub], false, false);
            else if (g == 1)
                accC[sub] = __builtin_amdgcn_wmma_f32_16x16x32_bf16(
                    false, a.v, false, bv, (short)0, accC[sub], false, false);
            else
                accX[sub] = __builtin_amdgcn_wmma_f32_16x16x32_bf16(
                    false, a.v, false, bv, (short)0, accX[sub], false, false);
        }
        // enforce: DSx6 (A frag + frags 0,1), 10 x (WMMA, DSx2), WMMAx2
        SGB(0x100, 6);
#pragma unroll
        for (int p = 0; p < 10; ++p) { SGB(0x008, 1); SGB(0x100, 2); }
        SGB(0x008, 2);

        wait_async0();
        __syncthreads();
    }

    // epilogue: add biases, Bx = B*x (f32), C (bf16)
#pragma unroll
    for (int sub = 0; sub < 4; ++sub) {
        int ng = n0 + wn * 64 + sub * 16 + l15;
        float bB = b_in[ng];
        float bC = b_in[HDIM + ng];
        float bx = b_in[2 * HDIM + ng];
#pragma unroll
        for (int r = 0; r < 8; ++r) {
            int tg = t0 + wm * 16 + khalf * 8 + r;
            float Bv = accB[sub][r] + bB;
            float xv = accX[sub][r] + bx;
            float Cv = accC[sub][r] + bC;
            Bx[(size_t)tg * HDIM + ng] = Bv * xv;
            Cg[(size_t)tg * HDIM + ng] = f2bf(Cv);
        }
    }
}

// segment-masked causal depthwise conv + output gate; y stored as bf16 (GEMM2 A operand)
__global__ void __launch_bounds__(256)
conv_gate_kernel(const float* __restrict__ Bx,          // [T][H]
                 const unsigned short* __restrict__ Cg, // [T][H]
                 const int* __restrict__ seg,           // [T]
                 const float* __restrict__ conv_w,      // [H][K]
                 const float* __restrict__ conv_b,      // [H]
                 unsigned short* __restrict__ Yg)       // [T][H] bf16
{
    int t = blockIdx.y;
    int h = (blockIdx.x * blockDim.x + threadIdx.x) * 4;
    if (h >= HDIM) return;
    int s = seg[t];

    const float4* cb4 = (const float4*)conv_b;
    const float4* cw4 = (const float4*)conv_w;   // row h -> all K taps contiguous
    float4 acc = cb4[h >> 2];
    float4 w0 = cw4[h + 0];
    float4 w1 = cw4[h + 1];
    float4 w2 = cw4[h + 2];
    float4 w3 = cw4[h + 3];

#pragma unroll
    for (int j = 0; j < KCONV; ++j) {   // tap j reads row t-j, weight col K-1-j
        int ts = t - j;
        if (ts >= 0 && seg[ts] == s) {
            float4 bx = *(const float4*)(Bx + (size_t)ts * HDIM + h);
            float t0w = (j == 0) ? w0.w : (j == 1) ? w0.z : (j == 2) ? w0.y : w0.x;
            float t1w = (j == 0) ? w1.w : (j == 1) ? w1.z : (j == 2) ? w1.y : w1.x;
            float t2w = (j == 0) ? w2.w : (j == 1) ? w2.z : (j == 2) ? w2.y : w2.x;
            float t3w = (j == 0) ? w3.w : (j == 1) ? w3.z : (j == 2) ? w3.y : w3.x;
            acc.x += t0w * bx.x;
            acc.y += t1w * bx.y;
            acc.z += t2w * bx.z;
            acc.w += t3w * bx.w;
        }
    }
    size_t base = (size_t)t * HDIM + h;
    Yg[base + 0] = f2bf(bf2f(Cg[base + 0]) * acc.x);
    Yg[base + 1] = f2bf(bf2f(Cg[base + 1]) * acc.y);
    Yg[base + 2] = f2bf(bf2f(Cg[base + 2]) * acc.z);
    Yg[base + 3] = f2bf(bf2f(Cg[base + 3]) * acc.w);
}

// GEMM2: out = y @ W_out + b_out   (f32 output)
__global__ void __launch_bounds__(256)
gemm2_kernel(const unsigned short* __restrict__ Yg,     // [T][H] bf16
             const unsigned short* __restrict__ WoutT,  // [H][H] bf16 (N-major)
             const float* __restrict__ b_out,           // [H]
             float* __restrict__ out)                   // [T][H] f32
{
    __shared__ unsigned short As[2][TM * LDA];
    __shared__ unsigned short Bs[2][TN * LDB];

    const int tid   = threadIdx.x;
    const int lane  = tid & 31;
    const int wave  = tid >> 5;
    const int wm    = wave >> 1;
    const int wn    = wave & 1;
    const int khalf = lane >> 4;
    const int l15   = lane & 15;
    const int t0    = blockIdx.y * TM;
    const int n0    = blockIdx.x * TN;

    v8f acc[4];
#pragma unroll
    for (int s = 0; s < 4; ++s) acc[s] = 0.0f;

    auto stage_load = [&](int k0, int s) {
        {
            int row = tid >> 2, ch = tid & 3;
            async_copy_b128(&As[s][row * LDA + ch * 8],
                            Yg + (size_t)(t0 + row) * HDIM + k0 + ch * 8);
        }
#pragma unroll
        for (int rep = 0; rep < 2; ++rep) {            // 128*4 = 512 chunks / 256 thr
            int i = rep * 256 + tid;
            int row = i >> 2, ch = i & 3;
            async_copy_b128(&Bs[s][row * LDB + ch * 8],
                            WoutT + (size_t)(n0 + row) * HDIM + k0 + ch * 8);
        }
    };

    stage_load(0, 0);
    wait_async0();
    __syncthreads();

    auto bfrag_addr = [&](int sub, int cur) -> const unsigned short* {
        const int n = wn * 64 + sub * 16 + l15;
        return &Bs[cur][n * LDB + khalf * 8];
    };

    const int NSTEP = HDIM / TK;
    for (int ks = 0; ks < NSTEP; ++ks) {
        const int cur = ks & 1;
        if (ks + 1 < NSTEP) stage_load((ks + 1) * TK, cur ^ 1);

        FragAB a;
        {
            const unsigned short* ap = &As[cur][(wm * 16 + l15) * LDA + khalf * 8];
            a.u[0] = *(const u32x4*)ap;
            a.u[1] = *(const u32x4*)(ap + 16);
        }
        FragAB bf[3];
        load_bfrag(bf[0], bfrag_addr(0, cur));
        load_bfrag(bf[1], bfrag_addr(1, cur));
#pragma unroll
        for (int sub = 0; sub < 4; ++sub) {
            if (sub + 2 < 4)
                load_bfrag(bf[(sub + 2) % 3], bfrag_addr(sub + 2, cur));
            acc[sub] = __builtin_amdgcn_wmma_f32_16x16x32_bf16(
                false, a.v, false, bf[sub % 3].v, (short)0, acc[sub], false, false);
        }
        // enforce: DSx6 (A frag + frags 0,1), 2 x (WMMA, DSx2), WMMAx2
        SGB(0x100, 6);
#pragma unroll
        for (int p = 0; p < 2; ++p) { SGB(0x008, 1); SGB(0x100, 2); }
        SGB(0x008, 2);

        wait_async0();
        __syncthreads();
    }

#pragma unroll
    for (int sub = 0; sub < 4; ++sub) {
        int ng = n0 + wn * 64 + sub * 16 + l15;
        float bo = b_out[ng];
#pragma unroll
        for (int r = 0; r < 8; ++r) {
            int tg = t0 + wm * 16 + khalf * 8 + r;
            out[(size_t)tg * HDIM + ng] = acc[sub][r] + bo;
        }
    }
}

// ---------------- launcher ----------------
extern "C" void kernel_launch(void* const* d_in, const int* in_sizes, int n_in,
                              void* d_out, int out_size, void* d_ws, size_t ws_size,
                              hipStream_t stream) {
    const float* hidden = (const float*)d_in[0];
    const int*   seg    = (const int*)d_in[1];
    const float* W_in   = (const float*)d_in[2];
    const float* b_in   = (const float*)d_in[3];
    const float* conv_w = (const float*)d_in[4];
    const float* conv_b = (const float*)d_in[5];
    const float* W_out  = (const float*)d_in[6];
    const float* b_out  = (const float*)d_in[7];
    float* out = (float*)d_out;

    char* ws = (char*)d_ws;
    size_t off = 0;
    auto alloc = [&](size_t bytes) -> void* {
        void* p = ws + off;
        off = (off + bytes + 255) & ~(size_t)255;
        return p;
    };
    unsigned short* Xb    = (unsigned short*)alloc((size_t)TDIM * HDIM * 2);      // bf16 X
    unsigned short* WinT  = (unsigned short*)alloc((size_t)3 * HDIM * HDIM * 2);  // bf16 W_in^T
    unsigned short* WoutT = (unsigned short*)alloc((size_t)HDIM * HDIM * 2);      // bf16 W_out^T
    float*          Bx    = (float*)alloc((size_t)TDIM * HDIM * 4);               // f32 gate product
    unsigned short* Cg    = (unsigned short*)alloc((size_t)TDIM * HDIM * 2);      // bf16 C gate
    unsigned short* Yg    = (unsigned short*)alloc((size_t)TDIM * HDIM * 2);      // bf16 y

    // 1) precision conversion (one-time, bandwidth-trivial)
    {
        int n = TDIM * HDIM;
        cvt_f32_to_bf16<<<(n + 255) / 256, 256, 0, stream>>>(hidden, Xb, n);
        transpose_to_bf16<<<dim3(3 * HDIM / 256, HDIM), 256, 0, stream>>>(W_in, WinT, HDIM, 3 * HDIM);
        transpose_to_bf16<<<dim3(HDIM / 256, HDIM), 256, 0, stream>>>(W_out, WoutT, HDIM, HDIM);
    }
    // 2) fused GEMM1 + input gate (async-staged, double-buffered, bf16 WMMA)
    gemm1_kernel<<<dim3(HDIM / TN, TDIM / TM), 256, 0, stream>>>(Xb, WinT, b_in, Bx, Cg);
    // 3) segment-masked causal conv + output gate
    conv_gate_kernel<<<dim3(HDIM / (4 * 256), TDIM), 256, 0, stream>>>(Bx, Cg, seg, conv_w, conv_b, Yg);
    // 4) GEMM2 + bias -> f32 output
    gemm2_kernel<<<dim3(HDIM / TN, TDIM / TM), 256, 0, stream>>>(Yg, WoutT, b_out, out);
}
